// FeedBack_953482740249
// MI455X (gfx1250) — matrix operand order repaired and checked by reading, not compiled, not run on hardware
//
#include <hip/hip_runtime.h>

// ---------------------------------------------------------------------------
// Persistent-tile LSTM encoder/decoder for MI455X (gfx1250, wave32, WMMA).
//   B=1024, T=128, F=128, U=512, S=96
// One workgroup (256 threads = 8 waves) owns 16 batch rows and runs all
// 128 warmup + 96 decode steps, with h (bf16, double buffered) and c (f32)
// resident in LDS. Weights are pre-packed to bf16 in WMMA B-fragment order
// and software-pipelined through a 2-deep register double buffer so the
// v_wmma_f32_16x16x32_bf16 stream never stalls on L2 latency.
// ---------------------------------------------------------------------------

typedef __attribute__((ext_vector_type(16))) __bf16 v16bf;
typedef __attribute__((ext_vector_type(8)))  __bf16 v8bf;
typedef __attribute__((ext_vector_type(8)))  float  v8f;

#define B_  1024
#define T_  128
#define F_  128
#define U_  512
#define S_  96

__device__ __forceinline__ float fast_rcp(float x) { return __builtin_amdgcn_rcpf(x); }
__device__ __forceinline__ float sigf(float x)     { return fast_rcp(1.0f + __expf(-x)); }
// tanh(x) = 1 - 2/(e^{2x}+1): one TRANS exp + one TRANS rcp (co-executes with WMMA)
__device__ __forceinline__ float tanh_f(float x)   { return 1.0f - 2.0f * fast_rcp(__expf(2.0f * x) + 1.0f); }

// A-matrix fragment (16x32 bf16) from a row-major [16 x stride] bf16 buffer.
// ISA layout: lane m=lane&15, kbase = 8*(lane>>4); elements 0..7 = K kbase+0..7,
// elements 8..15 = K kbase+16..23 (relative to k0).
__device__ __forceinline__ v16bf load_a(const __bf16* base, int stride, int k0, int lane) {
    const __bf16* p = base + (lane & 15) * stride + k0 + ((lane >> 4) << 3);
    v8bf lo = *(const v8bf*)(p);
    v8bf hi = *(const v8bf*)(p + 16);
    return __builtin_shufflevector(lo, hi, 0,1,2,3,4,5,6,7,8,9,10,11,12,13,14,15);
}

// B-matrix fragment (32x16 bf16) from the pre-swizzled weight image:
// packed[((nt*KT + kt)*32 + lane)*16 + j], 32 contiguous bytes per lane.
__device__ __forceinline__ v16bf load_b(const __bf16* pk, int KT, int nt, int kt, int lane) {
    return *(const v16bf*)(pk + ((((size_t)nt * KT + kt) << 5) + (size_t)lane) * 16);
}

__device__ __forceinline__ v8f wmma_bf(v16bf a, v16bf b, v8f c) {
    return __builtin_amdgcn_wmma_f32_16x16x32_bf16(
        /*neg_a=*/false, a, /*neg_b=*/false, b,
        /*c_mod=*/(short)0, c, /*reuse_a=*/false, /*reuse_b=*/false);
}

// One LSTM cell step for the 16-row tile. Reads xs (16xF_ bf16) and hcur
// (16xU_ bf16); writes hnxt (bf16) and cs (f32, wave-private columns).
// B fragments are double-buffered in registers (pipeline depth 1) across
// both the Wk and Wr K-loops, including the boundary between them.
__device__ __forceinline__ void lstm_step(const __bf16* xs, const __bf16* hcur,
                                          __bf16* hnxt, float* cs, const float* bs,
                                          const __bf16* WkP, const __bf16* WrP,
                                          int wave, int lane) {
    for (int j = 0; j < 4; ++j) {
        const int ug   = (wave << 2) + j;          // u-tile 0..31
        const int ncol = (ug << 4) + (lane & 15);  // column within U
        v8f acc[4] = {};                           // zero C -> inline-0 SRC2
        v16bf bb[2][4];                            // double-buffered B fragments

#pragma unroll
        for (int g = 0; g < 4; ++g) bb[0][g] = load_b(WkP, 4, g * 32 + ug, 0, lane);

        // x @ Wk  (K = 128 -> 4 k-tiles); prefetch next kt's B (Wr kt=0 at end)
#pragma unroll
        for (int kt = 0; kt < 4; ++kt) {
            const int cur = kt & 1, nxt = cur ^ 1;
#pragma unroll
            for (int g = 0; g < 4; ++g)
                bb[nxt][g] = (kt < 3) ? load_b(WkP, 4,  g * 32 + ug, kt + 1, lane)
                                      : load_b(WrP, 16, g * 32 + ug, 0,      lane);
            v16bf a = load_a(xs, F_, kt * 32, lane);
            acc[0] = wmma_bf(a, bb[cur][0], acc[0]);
            acc[1] = wmma_bf(a, bb[cur][1], acc[1]);
            acc[2] = wmma_bf(a, bb[cur][2], acc[2]);
            acc[3] = wmma_bf(a, bb[cur][3], acc[3]);
        }
        // h @ Wr  (K = 512 -> 16 k-tiles); buffer 0 holds kt=0 from above
#pragma unroll
        for (int kt = 0; kt < 16; ++kt) {
            const int cur = kt & 1, nxt = cur ^ 1;
            if (kt < 15) {
#pragma unroll
                for (int g = 0; g < 4; ++g)
                    bb[nxt][g] = load_b(WrP, 16, g * 32 + ug, kt + 1, lane);
            }
            v16bf a = load_a(hcur, U_, kt * 32, lane);
            acc[0] = wmma_bf(a, bb[cur][0], acc[0]);
            acc[1] = wmma_bf(a, bb[cur][1], acc[1]);
            acc[2] = wmma_bf(a, bb[cur][2], acc[2]);
            acc[3] = wmma_bf(a, bb[cur][3], acc[3]);
        }

        // Gate nonlinearities + state update. D layout: element r -> row
        // m = r + 8*(lane>>4), col = lane&15 within the tile.
        const float bi = bs[0 * U_ + ncol];
        const float bf = bs[1 * U_ + ncol];
        const float bg = bs[2 * U_ + ncol];
        const float bo = bs[3 * U_ + ncol];
#pragma unroll
        for (int r = 0; r < 8; ++r) {
            const int m   = r + ((lane >> 4) << 3);
            const int idx = m * U_ + ncol;
            float zi = acc[0][r] + bi;
            float zf = acc[1][r] + bf;
            float zg = acc[2][r] + bg;
            float zo = acc[3][r] + bo;
            float cN = sigf(zf) * cs[idx] + sigf(zi) * tanh_f(zg);
            cs[idx]  = cN;                       // c is wave-private: no barrier
            hnxt[idx] = (__bf16)(sigf(zo) * tanh_f(cN));
        }
    }
}

// pred = h @ Wd + bd : each of the 8 waves produces one 16-column tile of F.
// Writes fp32 to out[b, s, :] and bf16 to xs (next decode input).
__device__ __forceinline__ void pred_step(const __bf16* hcur, __bf16* xs,
                                          float* __restrict__ out, const float* bds,
                                          const __bf16* WdP, int wave, int lane,
                                          int b0, int s) {
    const int nt   = wave;                     // 8 tiles x 16 = 128 columns
    const int ncol = (nt << 4) + (lane & 15);
    v8f acc = {};
    v16bf bb[2];
    bb[0] = load_b(WdP, 16, nt, 0, lane);
#pragma unroll
    for (int kt = 0; kt < 16; ++kt) {
        const int cur = kt & 1, nxt = cur ^ 1;
        if (kt < 15) bb[nxt] = load_b(WdP, 16, nt, kt + 1, lane);
        v16bf a = load_a(hcur, U_, kt * 32, lane);
        acc = wmma_bf(a, bb[cur], acc);
    }
    const float bv = bds[ncol];
#pragma unroll
    for (int r = 0; r < 8; ++r) {
        const int m = r + ((lane >> 4) << 3);
        float p = acc[r] + bv;
        out[(size_t)(b0 + m) * (S_ * F_) + (size_t)s * F_ + ncol] = p;
        xs[m * F_ + ncol] = (__bf16)p;
    }
}

__global__ __launch_bounds__(256) void lstm_ar_kernel(
    const float* __restrict__ inputs, const float* __restrict__ bias4u,
    const float* __restrict__ bd, const __bf16* __restrict__ WkP,
    const __bf16* __restrict__ WrP, const __bf16* __restrict__ WdP,
    float* __restrict__ out) {
    __shared__ __attribute__((aligned(32))) __bf16 hs0[16 * U_];
    __shared__ __attribute__((aligned(32))) __bf16 hs1[16 * U_];
    __shared__ __attribute__((aligned(32))) __bf16 xs [16 * F_];
    __shared__ float cs[16 * U_];
    __shared__ float bs[4 * U_];
    __shared__ float bds[F_];

    const int tid  = threadIdx.x;
    const int lane = tid & 31;
    const int wave = tid >> 5;
    const int b0   = blockIdx.x << 4;   // 16 batch rows per workgroup

    for (int i = tid; i < 16 * U_; i += 256) { hs0[i] = (__bf16)0.0f; cs[i] = 0.0f; }
    for (int i = tid; i < 4 * U_;  i += 256) bs[i]  = bias4u[i];
    for (int i = tid; i < F_;      i += 256) bds[i] = bd[i];

    __bf16* hcur = hs0;
    __bf16* hnxt = hs1;

    // ---- warmup: scan over the input sequence ----
    for (int t = 0; t < T_; ++t) {
        // stage x_t tile (fp32 global -> bf16 LDS), coalesced
#pragma unroll
        for (int q = 0; q < 8; ++q) {
            int idx = (q << 8) + tid;                 // 0..2047
            int row = idx >> 7, col = idx & (F_ - 1);
            xs[idx] = (__bf16)inputs[(size_t)(b0 + row) * (T_ * F_) + (size_t)t * F_ + col];
        }
        // prefetch next step's x tile (one cacheline per participating thread)
        if (t + 1 < T_ && tid < 64) {
            int row = tid >> 2, seg = tid & 3;        // 16 rows x 4 x 32-float lines
            __builtin_prefetch(
                &inputs[(size_t)(b0 + row) * (T_ * F_) + (size_t)(t + 1) * F_ + seg * 32],
                0, 1);
        }
        __syncthreads();
        lstm_step(xs, hcur, hnxt, cs, bs, WkP, WrP, wave, lane);
        __syncthreads();
        __bf16* tmp = hcur; hcur = hnxt; hnxt = tmp;
    }

    // ---- autoregressive decode ----
    for (int s = 0; s < S_; ++s) {
        pred_step(hcur, xs, out, bds, WdP, wave, lane, b0, s);
        __syncthreads();
        if (s + 1 < S_) {
            lstm_step(xs, hcur, hnxt, cs, bs, WkP, WrP, wave, lane);
            __syncthreads();
            __bf16* tmp = hcur; hcur = hnxt; hnxt = tmp;
        }
    }
}

// Pre-swizzle a row-major fp32 weight [Ktot x Ntot] into bf16 WMMA B-fragment
// order: P[((nt*KT + kt)*32 + lane)*16 + j] = W[kt*32 + 8*(lane>>4) + (j&7) + 16*(j>>3)][nt*16 + (lane&15)]
__global__ void pack_wmma_b(const float* __restrict__ W, __bf16* __restrict__ P,
                            int Ktot, int Ntot) {
    int idx = blockIdx.x * blockDim.x + threadIdx.x;
    int total = Ktot * Ntot;
    if (idx >= total) return;
    int KT  = Ktot >> 5;
    int j   = idx & 15;
    int l   = (idx >> 4) & 31;
    int kt  = (idx >> 9) % KT;
    int nt  = (idx >> 9) / KT;
    int kk  = ((l >> 4) << 3) + (j & 7) + ((j >> 3) << 4);
    int col = (nt << 4) + (l & 15);
    P[idx] = (__bf16)W[(size_t)((kt << 5) + kk) * Ntot + col];
}

extern "C" void kernel_launch(void* const* d_in, const int* in_sizes, int n_in,
                              void* d_out, int out_size, void* d_ws, size_t ws_size,
                              hipStream_t stream) {
    (void)in_sizes; (void)n_in; (void)out_size; (void)ws_size;
    const float* inputs = (const float*)d_in[0];   // [B, T, F]
    const float* Wk     = (const float*)d_in[1];   // [F, 4U]
    const float* Wr     = (const float*)d_in[2];   // [U, 4U]
    const float* b      = (const float*)d_in[3];   // [4U]
    const float* Wd     = (const float*)d_in[4];   // [U, F]
    const float* bd     = (const float*)d_in[5];   // [F]

    __bf16* WkP = (__bf16*)d_ws;                    // 128*2048  bf16
    __bf16* WrP = WkP + (size_t)F_ * 4 * U_;        // 512*2048  bf16
    __bf16* WdP = WrP + (size_t)U_ * 4 * U_;        // 512*128   bf16

    pack_wmma_b<<<(F_ * 4 * U_ + 255) / 256, 256, 0, stream>>>(Wk, WkP, F_, 4 * U_);
    pack_wmma_b<<<(U_ * 4 * U_ + 255) / 256, 256, 0, stream>>>(Wr, WrP, U_, 4 * U_);
    pack_wmma_b<<<(U_ * F_    + 255) / 256, 256, 0, stream>>>(Wd, WdP, U_, F_);

    lstm_ar_kernel<<<B_ / 16, 256, 0, stream>>>(inputs, b, bd, WkP, WrP, WdP,
                                                (float*)d_out);
}
